// SCEncoder_51170240364774
// MI455X (gfx1250) — matrix operand order repaired and checked by reading, not compiled, not run on hardware
//
#include <hip/hip_runtime.h>

// ---------------------------------------------------------------------------
// Performer (FAVOR+) encoder for MI455X / gfx1250.
// GEMMs: v_wmma_f32_16x16x32_f16, 256x128x32 LDS-staged tiles (64x64 per
// wave => 16 WMMA per 16 ds_load_b128), tiles staged with
// global_load_async_to_lds_b128 (ASYNCcnt) instead of VGPR round-trips.
// f32 accumulation, fused bias/residual/GELU epilogues. Feature-map and
// linear-attention reductions (rank M=110) are VALU kernels (tiny FLOPs,
// L2-resident state). All scratch lives in d_ws (~0.5 GB needed).
// ---------------------------------------------------------------------------

#define B_    4
#define G_    16906
#define NSEQ  16907            // G_ + 1 (cls token)
#define NR    67628            // B_ * NSEQ
#define D_    256
#define H_    8
#define DH_   32
#define M_    110
#define L_    6
#define FF_   1024
#define EPS_  1e-4f

#define XSCALE 0.4204482076268573f    // DH^-0.25
#define XS2    0.17677669529663687f   // DH^-0.5
#define MSCALE 0.09534625892455922f   // M^-0.5
#define NEGBIG (-3.402823466e38f)
#define ENC_NEGINF 0x007FFFFFu        // fenc(-inf)

typedef _Float16 v16h __attribute__((ext_vector_type(16)));
typedef _Float16 v8h  __attribute__((ext_vector_type(8)));
typedef float    v8f  __attribute__((ext_vector_type(8)));
union HV { v16h v; v8h h[2]; };

// order-preserving float <-> uint encoding for atomicMax on floats
__device__ __forceinline__ unsigned fenc(float f) {
  unsigned b = __float_as_uint(f);
  return (b & 0x80000000u) ? ~b : (b | 0x80000000u);
}
__device__ __forceinline__ float fdec(unsigned u) {
  unsigned b = (u & 0x80000000u) ? (u & 0x7FFFFFFFu) : ~u;
  return __uint_as_float(b);
}

// async 16B copy global -> LDS (CDNA5 VGLOBAL async op, tracked by ASYNCcnt)
__device__ __forceinline__ void async_ld_b128(unsigned lds_off, const void* gp) {
  asm volatile("global_load_async_to_lds_b128 %0, %1, off"
               :: "v"(lds_off), "v"((unsigned long long)gp) : "memory");
}
__device__ __forceinline__ void wait_async0() {
  asm volatile("s_wait_asynccnt 0" ::: "memory");
}

// ---------------------------------------------------------------------------
// generic fill
__global__ void k_fill_u32(unsigned* p, unsigned v, int n) {
  int i = blockIdx.x * blockDim.x + threadIdx.x;
  if (i < n) p[i] = v;
}

// ---------------------------------------------------------------------------
// per-batch sum(|expr|) with 1e-12 floor
__global__ void k_rowsum_abs(const float* __restrict__ expr, float* __restrict__ sums) {
  __shared__ float red[256];
  int b = blockIdx.x, tid = threadIdx.x;
  float a = 0.f;
  for (int i = tid; i < G_; i += 256) a += fabsf(expr[(size_t)b * G_ + i]);
  red[tid] = a; __syncthreads();
  for (int o = 128; o > 0; o >>= 1) { if (tid < o) red[tid] += red[tid + o]; __syncthreads(); }
  if (tid == 0) sums[b] = fmaxf(red[0], 1e-12f);
}

// global max of e = log1p(expr/sum * 1e4)
__global__ void k_emax(const float* __restrict__ expr, const float* __restrict__ sums,
                       unsigned* emax_enc) {
  __shared__ float red[256];
  int idx = blockIdx.x * 256 + threadIdx.x;
  float e = NEGBIG;
  if (idx < B_ * G_) {
    int b = idx / G_;
    e = log1pf(expr[idx] / sums[b] * 1e4f);
  }
  red[threadIdx.x] = e; __syncthreads();
  for (int o = 128; o > 0; o >>= 1) {
    if (threadIdx.x < o) red[threadIdx.x] = fmaxf(red[threadIdx.x], red[threadIdx.x + o]);
    __syncthreads();
  }
  if (threadIdx.x == 0) atomicMax(emax_enc, fenc(red[0]));
}

// bucketize + embed: x[b][0] = cls, x[b][1+g] = token_emb[bin(e)]
__global__ void k_tokenize(const float* __restrict__ expr, const float* __restrict__ sums,
                           const unsigned* __restrict__ emax_enc,
                           const float* __restrict__ temb, const float* __restrict__ ctok,
                           float* __restrict__ x) {
  int blk = blockIdx.x;                 // 0 .. NR-1
  int b = blk / NSEQ, pos = blk % NSEQ;
  int tid = threadIdx.x;                // 0 .. 255
  size_t o = (size_t)blk * D_ + tid;
  if (pos == 0) { x[o] = ctok[tid]; return; }
  float emax = fdec(*emax_enc);
  float e = log1pf(expr[(size_t)b * G_ + (pos - 1)] / sums[b] * 1e4f);
  int t = 0;
  #pragma unroll
  for (int i = 0; i < 7; i++) t += ((emax * (float)i * (1.0f / 7.0f)) < e) ? 1 : 0;
  if (t > 6) t = 6;
  x[o] = temb[t * D_ + tid];
}

// ---------------------------------------------------------------------------
// LayerNorm f32 -> f16 (one row of D_=256 per block)
__global__ void k_layernorm(const float* __restrict__ xin, const float* __restrict__ g,
                            const float* __restrict__ bb, _Float16* __restrict__ out) {
  __shared__ float s1[256], s2[256];
  int row = blockIdx.x, tid = threadIdx.x;
  float v = xin[(size_t)row * D_ + tid];
  s1[tid] = v; s2[tid] = v * v; __syncthreads();
  for (int o = 128; o > 0; o >>= 1) {
    if (tid < o) { s1[tid] += s1[tid + o]; s2[tid] += s2[tid + o]; }
    __syncthreads();
  }
  float mu = s1[0] * (1.0f / D_);
  float var = s2[0] * (1.0f / D_) - mu * mu;
  float r = rsqrtf(var + 1e-5f);
  out[(size_t)row * D_ + tid] = (_Float16)((v - mu) * r * g[tid] + bb[tid]);
}

// ---------------------------------------------------------------------------
// weight transpose + f32->f16: src [K][N] -> dst [N][K]
__global__ void k_transpose(const float* __restrict__ src, _Float16* __restrict__ dst,
                            int K, int N) {
  int i = blockIdx.x * 256 + threadIdx.x;
  if (i >= N * K) return;
  int n = i / K, k = i % K;
  dst[i] = (_Float16)src[(size_t)k * N + n];
}

// ---------------------------------------------------------------------------
// WMMA GEMM: out[rows x N] (+)= A[rows x K] * Bt[N x K]^T + bias
// MODE 0: f32 store   MODE 1: f32 residual +=   MODE 2: f16 store of exact GELU
// Block tile 256x128x32, 8 waves as 4(M)x2(N), 64x64 per wave => 16 WMMA/step.
#define TM 256
#define TN 128
#define TK 32
#define LDH 40   // padded halves per LDS row (80B, 16B aligned, conflict-free-ish)

template<int MODE>
__global__ __launch_bounds__(256) void k_gemm(const _Float16* __restrict__ A,
                                              const _Float16* __restrict__ Bt,
                                              const float* __restrict__ bias,
                                              float* __restrict__ outf,
                                              _Float16* __restrict__ outh,
                                              int rows, int K, int N) {
  __shared__ _Float16 Asm[TM * LDH];
  __shared__ _Float16 Bsm[TN * LDH];
  const int tid = threadIdx.x;
  const int lane = tid & 31, wave = tid >> 5;
  const int wr = wave >> 1, wc = wave & 1;      // 4x2 wave grid -> 64x64 per wave
  const int hl = lane & 15, hi = lane >> 4;
  const int m0 = blockIdx.x * TM, n0 = blockIdx.y * TN;

  const unsigned lds_a = (unsigned)(unsigned long long)(&Asm[0]);
  const unsigned lds_b = (unsigned)(unsigned long long)(&Bsm[0]);

  v8f zero;
  #pragma unroll
  for (int e = 0; e < 8; e++) zero[e] = 0.0f;
  v8f acc[4][4];
  #pragma unroll
  for (int i = 0; i < 4; i++)
    #pragma unroll
    for (int j = 0; j < 4; j++) acc[i][j] = zero;

  const int ksteps = K / TK;
  for (int kt = 0; kt < ksteps; ++kt) {
    // async-stage A (256x32 halves) and B (128x32 halves) tiles into LDS.
    // Each thread: 4x 16B for A, 2x 16B for B, no VGPR round-trip.
    #pragma unroll
    for (int c = 0; c < 4; c++) {
      int idx = tid + c * 256;
      int r = idx >> 2;                  // 0..255
      int col = (idx & 3) << 3;          // 0,8,16,24
      int gr = m0 + r; if (gr >= rows) gr = rows - 1;
      async_ld_b128(lds_a + (unsigned)(r * LDH + col) * 2,
                    A + (size_t)gr * K + kt * TK + col);
    }
    #pragma unroll
    for (int c = 0; c < 2; c++) {
      int idx = tid + c * 256;
      int r = idx >> 2;                  // 0..127
      int col = (idx & 3) << 3;
      async_ld_b128(lds_b + (unsigned)(r * LDH + col) * 2,
                    Bt + (size_t)(n0 + r) * K + kt * TK + col);
    }
    wait_async0();
    __syncthreads();

    // A fragments (ISA 7.12.2: lanes<16 K=0..7 & 16..23, lanes>=16 K=8..15 & 24..31)
    v16h af[4];
    #pragma unroll
    for (int i = 0; i < 4; i++) {
      int row = wr * 64 + i * 16 + hl;
      int kb = hi * 8;
      HV u;
      u.h[0] = *(const v8h*)(&Asm[row * LDH + kb]);
      u.h[1] = *(const v8h*)(&Asm[row * LDH + kb + 16]);
      af[i] = u.v;
    }
    // B fragments (lanes<16 K=0..7 & 16..23, lanes>=16 K=8..15 & 24..31)
    v16h bf[4];
    #pragma unroll
    for (int j = 0; j < 4; j++) {
      int col = wc * 64 + j * 16 + hl;
      int kb = hi * 8;
      HV u;
      u.h[0] = *(const v8h*)(&Bsm[col * LDH + kb]);
      u.h[1] = *(const v8h*)(&Bsm[col * LDH + kb + 16]);
      bf[j] = u.v;
    }
    #pragma unroll
    for (int i = 0; i < 4; i++)
      #pragma unroll
      for (int j = 0; j < 4; j++)
        acc[i][j] = __builtin_amdgcn_wmma_f32_16x16x32_f16(
            false, af[i], false, bf[j], (short)0, acc[i][j], false, false);
    __syncthreads();
  }

  // epilogue (C layout: VGPR r -> M = r + 8*(lane>=16), N = lane&15)
  #pragma unroll
  for (int i = 0; i < 4; i++) {
    #pragma unroll
    for (int j = 0; j < 4; j++) {
      int col = n0 + wc * 64 + j * 16 + hl;
      float bb = bias[col];
      #pragma unroll
      for (int r = 0; r < 8; r++) {
        int row = m0 + wr * 64 + i * 16 + hi * 8 + r;
        if (row < rows) {
          float val = acc[i][j][r] + bb;
          size_t o = (size_t)row * N + col;
          if (MODE == 0) outf[o] = val;
          else if (MODE == 1) outf[o] += val;
          else outh[o] = (_Float16)(0.5f * val * (1.0f + erff(val * 0.70710678118654752f)));
        }
      }
    }
  }
}

// ---------------------------------------------------------------------------
// key stabilizer: global max over dd = XSCALE * (k_head . proj_m)
__global__ void k_kstab(const float* __restrict__ kbuf, const float* __restrict__ proj,
                        unsigned* dmax_enc) {
  __shared__ float kr[256];
  __shared__ float red[128];
  int row = blockIdx.x, tid = threadIdx.x;   // block = 128
  kr[tid]       = kbuf[(size_t)row * D_ + tid];
  kr[tid + 128] = kbuf[(size_t)row * D_ + tid + 128];
  __syncthreads();
  float mx = NEGBIG;
  if (tid < M_) {
    const float* pm = proj + tid * DH_;
    #pragma unroll
    for (int h = 0; h < H_; h++) {
      const float* kh = &kr[h * DH_];
      float dd = 0.f;
      #pragma unroll
      for (int d = 0; d < DH_; d++) dd += kh[d] * pm[d];
      mx = fmaxf(mx, dd * XSCALE);
    }
  }
  red[tid] = mx; __syncthreads();
  for (int o = 64; o > 0; o >>= 1) {
    if (tid < o) red[tid] = fmaxf(red[tid], red[tid + o]);
    __syncthreads();
  }
  if (tid == 0) atomicMax(dmax_enc, fenc(red[0]));
}

// kf on-the-fly -> accumulate ctx[b,h,m,d] and ksum[b,h,m]
#define CH 256
__global__ void k_ctx(const float* __restrict__ kbuf, const float* __restrict__ vbuf,
                      const float* __restrict__ proj, const unsigned* __restrict__ dmax_enc,
                      float* __restrict__ ctx, float* __restrict__ ksum) {
  __shared__ float kh[DH_], vh[DH_];
  __shared__ float dsh;
  int bh = blockIdx.y;                 // b*H + h
  int b = bh / H_, h = bh % H_;
  int tid = threadIdx.x;               // block = 128
  int i0 = blockIdx.x * CH;
  int i1 = i0 + CH; if (i1 > NSEQ) i1 = NSEQ;
  float dmax = fdec(*dmax_enc);
  float acc[DH_];
  #pragma unroll
  for (int d = 0; d < DH_; d++) acc[d] = 0.f;
  float asum = 0.f;

  for (int i = i0; i < i1; i++) {
    size_t base = ((size_t)(b * NSEQ + i)) * D_ + h * DH_;
    if (tid < DH_) kh[tid] = kbuf[base + tid];
    else if (tid < 2 * DH_) vh[tid - DH_] = vbuf[base + tid - DH_];
    __syncthreads();
    if (tid == 0) {
      float s = 0.f;
      #pragma unroll
      for (int d = 0; d < DH_; d++) s += kh[d] * kh[d];
      dsh = 0.5f * XS2 * s;
    }
    __syncthreads();
    if (tid < M_) {
      const float* pm = proj + tid * DH_;
      float dd = 0.f;
      #pragma unroll
      for (int d = 0; d < DH_; d++) dd += kh[d] * pm[d];
      float kf = MSCALE * (expf(dd * XSCALE - dsh - dmax) + EPS_);
      asum += kf;
      #pragma unroll
      for (int d = 0; d < DH_; d++) acc[d] += kf * vh[d];
    }
    __syncthreads();
  }
  if (tid < M_) {
    float* cp = ctx + ((size_t)bh * M_ + tid) * DH_;
    #pragma unroll
    for (int d = 0; d < DH_; d++) atomicAdd(cp + d, acc[d]);
    atomicAdd(ksum + (size_t)bh * M_ + tid, asum);
  }
}

// qf per row + o = (qf . ctx) / (qf . ksum), written f16 (input of wo GEMM)
__global__ void k_attno(const float* __restrict__ qbuf, const float* __restrict__ ctx,
                        const float* __restrict__ ksum, const float* __restrict__ proj,
                        _Float16* __restrict__ o16) {
  __shared__ float qh[DH_];
  __shared__ float qf[M_];
  __shared__ float red[128];
  int r = blockIdx.x / H_;
  int h = blockIdx.x % H_;
  int b = r / NSEQ;
  int bh = b * H_ + h;
  int tid = threadIdx.x;               // block = 128
  if (tid < DH_) qh[tid] = qbuf[(size_t)r * D_ + h * DH_ + tid];
  __syncthreads();
  float dd = NEGBIG;
  if (tid < M_) {
    const float* pm = proj + tid * DH_;
    float s = 0.f;
    #pragma unroll
    for (int d = 0; d < DH_; d++) s += qh[d] * pm[d];
    dd = s * XSCALE;
  }
  red[tid] = dd; __syncthreads();
  for (int o = 64; o > 0; o >>= 1) {
    if (tid < o) red[tid] = fmaxf(red[tid], red[tid + o]);
    __syncthreads();
  }
  float rowmax = red[0];
  __syncthreads();
  if (tid < M_) {
    float s = 0.f;
    #pragma unroll
    for (int d = 0; d < DH_; d++) s += qh[d] * qh[d];
    qf[tid] = MSCALE * (expf(dd - 0.5f * XS2 * s - rowmax) + EPS_);
  }
  __syncthreads();
  float part = 0.f;
  for (int m = tid; m < M_; m += 128) part += qf[m] * ksum[(size_t)bh * M_ + m];
  red[tid] = part; __syncthreads();
  for (int o = 64; o > 0; o >>= 1) {
    if (tid < o) red[tid] += red[tid + o];
    __syncthreads();
  }
  float dinv = 1.0f / red[0];
  if (tid < DH_) {
    const float* cp = ctx + (size_t)bh * M_ * DH_ + tid;
    float o = 0.f;
    for (int m = 0; m < M_; m++) o += qf[m] * cp[(size_t)m * DH_];
    o16[(size_t)r * D_ + h * DH_ + tid] = (_Float16)(o * dinv);
  }
}

// ---------------------------------------------------------------------------
// final projection: out[b] = x[b,0,:] @ proj_w + proj_b   (4 x 256)
__global__ void k_finalproj(const float* __restrict__ x, const float* __restrict__ pw,
                            const float* __restrict__ pb, float* __restrict__ out) {
  __shared__ float xs[D_];
  int b = blockIdx.x, tid = threadIdx.x;
  xs[tid] = x[((size_t)b * NSEQ) * D_ + tid];
  __syncthreads();
  float acc = pb[tid];
  for (int d = 0; d < D_; d++) acc += xs[d] * pw[(size_t)d * D_ + tid];
  out[(size_t)b * D_ + tid] = acc;
}

// ---------------------------------------------------------------------------
extern "C" void kernel_launch(void* const* d_in, const int* in_sizes, int n_in,
                              void* d_out, int out_size, void* d_ws, size_t ws_size,
                              hipStream_t stream) {
  (void)in_sizes; (void)n_in; (void)out_size; (void)ws_size;
  const float* expr  = (const float*)d_in[0];
  const float* temb  = (const float*)d_in[1];
  const float* ctok  = (const float*)d_in[2];
  const float* ln1g  = (const float*)d_in[3];
  const float* ln1b  = (const float*)d_in[4];
  const float* wq    = (const float*)d_in[5];
  const float* bq    = (const float*)d_in[6];
  const float* wk    = (const float*)d_in[7];
  const float* bk    = (const float*)d_in[8];
  const float* wv    = (const float*)d_in[9];
  const float* bv    = (const float*)d_in[10];
  const float* wo    = (const float*)d_in[11];
  const float* bo    = (const float*)d_in[12];
  const float* ln2g  = (const float*)d_in[13];
  const float* ln2b  = (const float*)d_in[14];
  const float* w1    = (const float*)d_in[15];
  const float* b1    = (const float*)d_in[16];
  const float* w2    = (const float*)d_in[17];
  const float* b2    = (const float*)d_in[18];
  const float* pw    = (const float*)d_in[19];
  const float* pb    = (const float*)d_in[20];
  const float* projs = (const float*)d_in[21];

  // ---- workspace layout ----
  char* wsp = (char*)d_ws;
  auto alloc = [&](size_t bytes) -> char* {
    char* p = wsp; wsp += (bytes + 255) & ~(size_t)255; return p;
  };
  float*    x    = (float*)   alloc((size_t)NR * D_ * 4);
  _Float16* h16  = (_Float16*)alloc((size_t)NR * D_ * 2);
  float*    q    = (float*)   alloc((size_t)NR * D_ * 4);
  float*    k    = (float*)   alloc((size_t)NR * D_ * 4);
  float*    v    = (float*)   alloc((size_t)NR * D_ * 4);
  _Float16* ff16 = (_Float16*)alloc((size_t)NR * FF_ * 2);
  float*    ctx  = (float*)   alloc((size_t)B_ * H_ * M_ * DH_ * 4);
  float*    ksum = (float*)   alloc((size_t)B_ * H_ * M_ * 4);
  _Float16* wt16 = (_Float16*)alloc((size_t)L_ * 786432 * 2);
  float*    sums = (float*)   alloc(B_ * 4);
  unsigned* emax = (unsigned*)alloc(4);
  unsigned* dmax = (unsigned*)alloc(4);

  const size_t LW = 786432;   // f16 halves per layer of prepped weights
  const size_t OWQ = 0, OWK = 65536, OWV = 131072, OWO = 196608,
               OW1 = 262144, OW2 = 524288;

  // ---- weight prep: transpose + f32->f16 ----
  for (int l = 0; l < L_; l++) {
    _Float16* base = wt16 + (size_t)l * LW;
    k_transpose<<<(D_ * D_ + 255) / 256, 256, 0, stream>>>(wq + (size_t)l * D_ * D_, base + OWQ, D_, D_);
    k_transpose<<<(D_ * D_ + 255) / 256, 256, 0, stream>>>(wk + (size_t)l * D_ * D_, base + OWK, D_, D_);
    k_transpose<<<(D_ * D_ + 255) / 256, 256, 0, stream>>>(wv + (size_t)l * D_ * D_, base + OWV, D_, D_);
    k_transpose<<<(D_ * D_ + 255) / 256, 256, 0, stream>>>(wo + (size_t)l * D_ * D_, base + OWO, D_, D_);
    k_transpose<<<(D_ * FF_ + 255) / 256, 256, 0, stream>>>(w1 + (size_t)l * D_ * FF_, base + OW1, D_, FF_);
    k_transpose<<<(D_ * FF_ + 255) / 256, 256, 0, stream>>>(w2 + (size_t)l * FF_ * D_, base + OW2, FF_, D_);
  }

  // ---- preprocess ----
  k_fill_u32<<<1, 32, 0, stream>>>(emax, ENC_NEGINF, 1);
  k_rowsum_abs<<<B_, 256, 0, stream>>>(expr, sums);
  k_emax<<<(B_ * G_ + 255) / 256, 256, 0, stream>>>(expr, sums, emax);
  k_tokenize<<<NR, 256, 0, stream>>>(expr, sums, emax, temb, ctok, x);

  const int rowTiles = (NR + TM - 1) / TM;   // 265

  // ---- layers ----
  for (int l = 0; l < L_; l++) {
    const _Float16* base = wt16 + (size_t)l * LW;
    const float* pj = projs + (size_t)l * M_ * DH_;

    k_layernorm<<<NR, 256, 0, stream>>>(x, ln1g + l * D_, ln1b + l * D_, h16);

    dim3 g2(rowTiles, D_ / TN);
    k_gemm<0><<<g2, 256, 0, stream>>>(h16, base + OWQ, bq + l * D_, q, nullptr, NR, D_, D_);
    k_gemm<0><<<g2, 256, 0, stream>>>(h16, base + OWK, bk + l * D_, k, nullptr, NR, D_, D_);
    k_gemm<0><<<g2, 256, 0, stream>>>(h16, base + OWV, bv + l * D_, v, nullptr, NR, D_, D_);

    k_fill_u32<<<1, 32, 0, stream>>>(dmax, ENC_NEGINF, 1);
    k_fill_u32<<<(B_ * H_ * M_ * DH_ + 255) / 256, 256, 0, stream>>>((unsigned*)ctx, 0u, B_ * H_ * M_ * DH_);
    k_fill_u32<<<(B_ * H_ * M_ + 255) / 256, 256, 0, stream>>>((unsigned*)ksum, 0u, B_ * H_ * M_);

    k_kstab<<<NR, 128, 0, stream>>>(k, pj, dmax);
    dim3 gc((NSEQ + CH - 1) / CH, B_ * H_);
    k_ctx<<<gc, 128, 0, stream>>>(k, v, pj, dmax, ctx, ksum);
    k_attno<<<NR * H_, 128, 0, stream>>>(q, ctx, ksum, pj, h16);

    k_gemm<1><<<g2, 256, 0, stream>>>(h16, base + OWO, bo + l * D_, x, nullptr, NR, D_, D_);

    k_layernorm<<<NR, 256, 0, stream>>>(x, ln2g + l * D_, ln2b + l * D_, h16);

    dim3 g1(rowTiles, FF_ / TN);
    k_gemm<2><<<g1, 256, 0, stream>>>(h16, base + OW1, b1 + l * FF_, nullptr, ff16, NR, D_, FF_);
    k_gemm<1><<<g2, 256, 0, stream>>>(ff16, base + OW2, b2 + l * D_, x, nullptr, NR, FF_, D_);
  }

  k_finalproj<<<B_, 256, 0, stream>>>(x, pw, pb, (float*)d_out);
}
